// MobileViTv4_block_85031762526427
// MI455X (gfx1250) — matrix operand order repaired and checked by reading, not compiled, not run on hardware
//
#include <hip/hip_runtime.h>
#include <hip/hip_bf16.h>
#include <math.h>

// ---------------- constants ----------------
#define BATCH 8
#define CH    96
#define HW    4096           // 64*64
#define DIMD  96
#define NTOK  32768          // B * 4 patches * 1024 tokens
#define INNER 32
#define MLPD  192
#define EPSV  1e-5f

typedef __attribute__((ext_vector_type(16))) __bf16 v16bf;
typedef __attribute__((ext_vector_type(8)))  float  v8f;

// ---------------- device helpers ----------------
__device__ __forceinline__ float silu_f(float v) {
    return v / (1.0f + __expf(-v));
}

__device__ __forceinline__ v8f wmma_bf16(v16bf a, v16bf b, v8f c) {
    // D = A(16x32 bf16) * B(32x16 bf16) + C(16x16 f32)
    return __builtin_amdgcn_wmma_f32_16x16x32_bf16(
        false, a, false, b, (short)0, c, false, false);
}

// A fragment: weight row-major [M][K], rows m = lane&15, lane-half selects K sub-block.
// Contiguous K order (matches B): lane half h holds K = 16*h + e, e = 0..15.
__device__ __forceinline__ v16bf load_frag_a(const __bf16* W, int m0, int k0, int ldk) {
    int lane = threadIdx.x & 31;
    int m = lane & 15, half = lane >> 4;
    const __bf16* p0 = W + (size_t)(m0 + m) * ldk + k0 + half * 16;
    const __bf16* p1 = p0 + 8;
    v16bf a;
#pragma unroll
    for (int e = 0; e < 8; ++e) a[e] = p0[e];
#pragma unroll
    for (int e = 0; e < 8; ++e) a[8 + e] = p1[e];
    return a;
}

// B fragment from a per-lane row pointer (activation [token][feat], unit stride in K).
// Contiguous K order (matches A): lanes 0-15 hold K=0..15, lanes 16-31 hold K=16..31 of the 32x16 tile.
__device__ __forceinline__ v16bf load_frag_b_bf16(const __bf16* row, int k0) {
    int khalf = (threadIdx.x & 31) >> 4;
    const __bf16* p = row + k0 + khalf * 16;
    v16bf b;
#pragma unroll
    for (int e = 0; e < 16; ++e) b[e] = p[e];
    return b;
}

__device__ __forceinline__ v16bf load_frag_b_f32(const float* row, int k0) {
    int khalf = (threadIdx.x & 31) >> 4;
    const float* p = row + k0 + khalf * 16;
    v16bf b;
#pragma unroll
    for (int e = 0; e < 16; ++e) b[e] = (__bf16)p[e];
    return b;
}

// ---------------- kernels ----------------

// f32 -> bf16 weight conversion
__global__ void cvt_bf16_kernel(const float* __restrict__ s, __bf16* __restrict__ d, int n) {
    int i = blockIdx.x * blockDim.x + threadIdx.x;
    if (i < n) d[i] = (__bf16)s[i];
}

// depthwise 3x3 + BN + SiLU
__global__ void dwconv_kernel(const float* __restrict__ x, const float* __restrict__ w,
                              const float* __restrict__ g, const float* __restrict__ bb,
                              const float* __restrict__ mm, const float* __restrict__ vv,
                              float* __restrict__ out) {
    int idx = blockIdx.x * blockDim.x + threadIdx.x;
    if (idx >= BATCH * CH * HW) return;
    int xw = idx & 63;
    int y  = (idx >> 6) & 63;
    int c  = (idx >> 12) % CH;
    int b  = idx / (CH * HW);
    const float* xp = x + ((size_t)(b * CH + c)) * HW;
    const float* wp = w + c * 9;
    float sum = 0.f;
#pragma unroll
    for (int dy = 0; dy < 3; ++dy) {
        int yy = y + dy - 1;
        if (yy < 0 || yy > 63) continue;
#pragma unroll
        for (int dx = 0; dx < 3; ++dx) {
            int xx = xw + dx - 1;
            if (xx < 0 || xx > 63) continue;
            sum += xp[yy * 64 + xx] * wp[dy * 3 + dx];
        }
    }
    float inv = g[c] * rsqrtf(vv[c] + EPSV);
    out[idx] = silu_f(sum * inv + (bb[c] - mm[c] * inv));
}

// pointwise conv (96->96) as WMMA GEMM + BN + SiLU, writes token-layout t[token][96].
// Each wave: 16 output channels x 64 pixels (4 B-tiles share each A fragment).
__global__ void pwconv_kernel(const __bf16* __restrict__ Wbf, const float* __restrict__ h1,
                              const float* __restrict__ g, const float* __restrict__ bb,
                              const float* __restrict__ mm, const float* __restrict__ vv,
                              float* __restrict__ T) {
    int wave  = (blockIdx.x * blockDim.x + threadIdx.x) >> 5;
    int mtile = wave % 6;
    int ntile = wave / 6;           // [0,512)
    int lane  = threadIdx.x & 31;
    int n = lane & 15, half = lane >> 4;
    __builtin_prefetch(Wbf + (size_t)mtile * 16 * 96, 0, 1);

    int pix[4], bimg[4], yx[4];
    const float* base[4];
#pragma unroll
    for (int j = 0; j < 4; ++j) {
        pix[j]  = ntile * 64 + j * 16 + n;
        bimg[j] = pix[j] >> 12;
        yx[j]   = pix[j] & 4095;
        base[j] = h1 + (size_t)bimg[j] * CH * HW + yx[j];
    }
    v8f z = {};
    v8f acc[4] = {z, z, z, z};
#pragma unroll
    for (int k0 = 0; k0 < 96; k0 += 32) {
        v16bf a = load_frag_a(Wbf, mtile * 16, k0, 96);
        int kb = k0 + half * 16;
#pragma unroll
        for (int j = 0; j < 4; ++j) {
            v16bf bf;
#pragma unroll
            for (int e = 0; e < 16; ++e) bf[e] = (__bf16)base[j][(size_t)(kb + e) * HW];
            acc[j] = wmma_bf16(a, bf, acc[j]);
        }
    }
    int d0 = mtile * 16 + half * 8;
#pragma unroll
    for (int j = 0; j < 4; ++j) {
        int y = yx[j] >> 6, xw = yx[j] & 63;
        int token = (bimg[j] * 4 + (y & 1) * 2 + (xw & 1)) * 1024
                  + ((y >> 1) * 32 + (xw >> 1));
        float* op = T + (size_t)token * DIMD + d0;
#pragma unroll
        for (int r = 0; r < 8; ++r) {
            int d = d0 + r;
            float inv = g[d] * rsqrtf(vv[d] + EPSV);
            op[r] = silu_f(acc[j][r] * inv + (bb[d] - mm[d] * inv));
        }
    }
}

// LayerNorm over D=96, one wave per token, bf16 output
__global__ void ln_kernel(const float* __restrict__ T, const float* __restrict__ g,
                          const float* __restrict__ b, __bf16* __restrict__ out) {
    int token = (blockIdx.x * blockDim.x + threadIdx.x) >> 5;
    int lane  = threadIdx.x & 31;
    const float* row = T + (size_t)token * DIMD;
    float x0 = row[lane], x1 = row[lane + 32], x2 = row[lane + 64];
    float s  = x0 + x1 + x2;
    float sq = x0 * x0 + x1 * x1 + x2 * x2;
#pragma unroll
    for (int m = 16; m >= 1; m >>= 1) {
        s  += __shfl_xor(s,  m, 32);
        sq += __shfl_xor(sq, m, 32);
    }
    float mean = s * (1.f / 96.f);
    float var  = sq * (1.f / 96.f) - mean * mean;
    float rstd = rsqrtf(var + EPSV);
    __bf16* orow = out + (size_t)token * DIMD;
    orow[lane]      = (__bf16)((x0 - mean) * rstd * g[lane]      + b[lane]);
    orow[lane + 32] = (__bf16)((x1 - mean) * rstd * g[lane + 32] + b[lane + 32]);
    orow[lane + 64] = (__bf16)((x2 - mean) * rstd * g[lane + 64] + b[lane + 64]);
}

// QKV: out[token][96] = xn[token][:] . Wqkv[e][:]   (16x64 per wave)
__global__ void qkv_kernel(const __bf16* __restrict__ Wbf, const __bf16* __restrict__ X,
                           float* __restrict__ out) {
    int wave  = (blockIdx.x * blockDim.x + threadIdx.x) >> 5;
    int mtile = wave % 6;
    int ntile = wave / 6;
    int lane  = threadIdx.x & 31;
    int n = lane & 15, half = lane >> 4;
    __builtin_prefetch(Wbf + (size_t)mtile * 16 * 96, 0, 1);
    const __bf16* xrow[4];
#pragma unroll
    for (int j = 0; j < 4; ++j)
        xrow[j] = X + (size_t)(ntile * 64 + j * 16 + n) * DIMD;
    v8f z = {};
    v8f acc[4] = {z, z, z, z};
#pragma unroll
    for (int k0 = 0; k0 < 96; k0 += 32) {
        v16bf a = load_frag_a(Wbf, mtile * 16, k0, 96);
#pragma unroll
        for (int j = 0; j < 4; ++j)
            acc[j] = wmma_bf16(a, load_frag_b_bf16(xrow[j], k0), acc[j]);
    }
    int e0 = mtile * 16 + half * 8;
#pragma unroll
    for (int j = 0; j < 4; ++j) {
        float* op = out + (size_t)(ntile * 64 + j * 16 + n) * DIMD + e0;
#pragma unroll
        for (int r = 0; r < 8; ++r) op[r] = acc[j][r];
    }
}

// attention: one block per (b,p,head); K/V staged in LDS; online softmax
__global__ void attn_kernel(const float* __restrict__ qkv, __bf16* __restrict__ obf) {
    extern __shared__ float smem[];
    float* kk = smem;            // 1024*8
    float* vv = smem + 8192;     // 1024*8
    int bp   = blockIdx.x >> 2;
    int head = blockIdx.x & 3;
    int tok0 = bp << 10;
    int tid  = threadIdx.x;
    for (int r = tid; r < 1024; r += 256) {
        const float* src = qkv + (size_t)(tok0 + r) * DIMD + head * 8;
#pragma unroll
        for (int d = 0; d < 8; ++d) {
            kk[r * 8 + d] = src[32 + d];
            vv[r * 8 + d] = src[64 + d];
        }
    }
    __syncthreads();
    const float scale = 0.35355339059327373f;  // 1/sqrt(8)
    for (int q = tid; q < 1024; q += 256) {
        const float* qs = qkv + (size_t)(tok0 + q) * DIMD + head * 8;
        float qv[8];
#pragma unroll
        for (int d = 0; d < 8; ++d) qv[d] = qs[d] * scale;
        float Mx = -INFINITY, S = 0.f;
        float acc[8] = {0.f, 0.f, 0.f, 0.f, 0.f, 0.f, 0.f, 0.f};
        for (int m = 0; m < 1024; ++m) {
            const float* kr = kk + m * 8;
            float s = qv[0]*kr[0] + qv[1]*kr[1] + qv[2]*kr[2] + qv[3]*kr[3]
                    + qv[4]*kr[4] + qv[5]*kr[5] + qv[6]*kr[6] + qv[7]*kr[7];
            float nm   = fmaxf(Mx, s);
            float corr = __expf(Mx - nm);
            float w    = __expf(s - nm);
            S = S * corr + w;
            const float* vr = vv + m * 8;
#pragma unroll
            for (int d = 0; d < 8; ++d) acc[d] = acc[d] * corr + w * vr[d];
            Mx = nm;
        }
        float rS = 1.f / S;
        __bf16* op = obf + (size_t)(tok0 + q) * INNER + head * 8;
#pragma unroll
        for (int d = 0; d < 8; ++d) op[d] = (__bf16)(acc[d] * rS);
    }
}

// out projection (32->96) + bias + residual into t   (16x64 per wave)
__global__ void proj_kernel(const __bf16* __restrict__ Wbf, const __bf16* __restrict__ O,
                            const float* __restrict__ bias, float* __restrict__ T) {
    int wave  = (blockIdx.x * blockDim.x + threadIdx.x) >> 5;
    int mtile = wave % 6;
    int ntile = wave / 6;
    int lane  = threadIdx.x & 31;
    int n = lane & 15, half = lane >> 4;
    __builtin_prefetch(Wbf + (size_t)mtile * 16 * INNER, 0, 1);
    v8f z = {};
    v8f acc[4] = {z, z, z, z};
    v16bf a = load_frag_a(Wbf, mtile * 16, 0, INNER);
#pragma unroll
    for (int j = 0; j < 4; ++j) {
        const __bf16* orow = O + (size_t)(ntile * 64 + j * 16 + n) * INNER;
        acc[j] = wmma_bf16(a, load_frag_b_bf16(orow, 0), acc[j]);
    }
    int d0 = mtile * 16 + half * 8;
#pragma unroll
    for (int j = 0; j < 4; ++j) {
        float* tp = T + (size_t)(ntile * 64 + j * 16 + n) * DIMD + d0;
#pragma unroll
        for (int r = 0; r < 8; ++r) tp[r] = tp[r] + acc[j][r] + bias[d0 + r];
    }
}

// FFN1: 96->192, SiLU, bf16 out   (16x64 per wave)
__global__ void ffn1_kernel(const __bf16* __restrict__ Wbf, const __bf16* __restrict__ X,
                            const float* __restrict__ bias, __bf16* __restrict__ hid) {
    int wave  = (blockIdx.x * blockDim.x + threadIdx.x) >> 5;
    int mtile = wave % 12;
    int ntile = wave / 12;
    int lane  = threadIdx.x & 31;
    int n = lane & 15, half = lane >> 4;
    __builtin_prefetch(Wbf + (size_t)mtile * 16 * 96, 0, 1);
    const __bf16* xrow[4];
#pragma unroll
    for (int j = 0; j < 4; ++j)
        xrow[j] = X + (size_t)(ntile * 64 + j * 16 + n) * DIMD;
    v8f z = {};
    v8f acc[4] = {z, z, z, z};
#pragma unroll
    for (int k0 = 0; k0 < 96; k0 += 32) {
        v16bf a = load_frag_a(Wbf, mtile * 16, k0, 96);
#pragma unroll
        for (int j = 0; j < 4; ++j)
            acc[j] = wmma_bf16(a, load_frag_b_bf16(xrow[j], k0), acc[j]);
    }
    int e0 = mtile * 16 + half * 8;
#pragma unroll
    for (int j = 0; j < 4; ++j) {
        __bf16* hp = hid + (size_t)(ntile * 64 + j * 16 + n) * MLPD + e0;
#pragma unroll
        for (int r = 0; r < 8; ++r) hp[r] = (__bf16)silu_f(acc[j][r] + bias[e0 + r]);
    }
}

// FFN2: 192->96 + bias + residual into t   (16x64 per wave)
__global__ void ffn2_kernel(const __bf16* __restrict__ Wbf, const __bf16* __restrict__ Hid,
                            const float* __restrict__ bias, float* __restrict__ T) {
    int wave  = (blockIdx.x * blockDim.x + threadIdx.x) >> 5;
    int mtile = wave % 6;
    int ntile = wave / 6;
    int lane  = threadIdx.x & 31;
    int n = lane & 15, half = lane >> 4;
    __builtin_prefetch(Wbf + (size_t)mtile * 16 * 192, 0, 1);
    const __bf16* hrow[4];
#pragma unroll
    for (int j = 0; j < 4; ++j)
        hrow[j] = Hid + (size_t)(ntile * 64 + j * 16 + n) * MLPD;
    v8f z = {};
    v8f acc[4] = {z, z, z, z};
#pragma unroll
    for (int k0 = 0; k0 < 192; k0 += 32) {
        v16bf a = load_frag_a(Wbf, mtile * 16, k0, 192);
#pragma unroll
        for (int j = 0; j < 4; ++j)
            acc[j] = wmma_bf16(a, load_frag_b_bf16(hrow[j], k0), acc[j]);
    }
    int d0 = mtile * 16 + half * 8;
#pragma unroll
    for (int j = 0; j < 4; ++j) {
        float* tp = T + (size_t)(ntile * 64 + j * 16 + n) * DIMD + d0;
#pragma unroll
        for (int r = 0; r < 8; ++r) tp[r] = tp[r] + acc[j][r] + bias[d0 + r];
    }
}

// fuse conv (96->96 over un-patched t) + BN + SiLU + outer residual  (16x64 per wave)
__global__ void fuse_kernel(const __bf16* __restrict__ Wbf, const float* __restrict__ T,
                            const float* __restrict__ g, const float* __restrict__ bb,
                            const float* __restrict__ mm, const float* __restrict__ vv,
                            const float* __restrict__ x, float* __restrict__ out) {
    int wave  = (blockIdx.x * blockDim.x + threadIdx.x) >> 5;
    int mtile = wave % 6;
    int ntile = wave / 6;
    int lane  = threadIdx.x & 31;
    int n = lane & 15, half = lane >> 4;
    __builtin_prefetch(Wbf + (size_t)mtile * 16 * 96, 0, 1);
    int bimg[4], yx[4];
    const float* trow[4];
#pragma unroll
    for (int j = 0; j < 4; ++j) {
        int pix = ntile * 64 + j * 16 + n;
        bimg[j] = pix >> 12;
        yx[j]   = pix & 4095;
        int y = yx[j] >> 6, xw = yx[j] & 63;
        int token = (bimg[j] * 4 + (y & 1) * 2 + (xw & 1)) * 1024
                  + ((y >> 1) * 32 + (xw >> 1));
        trow[j] = T + (size_t)token * DIMD;
    }
    v8f z = {};
    v8f acc[4] = {z, z, z, z};
#pragma unroll
    for (int k0 = 0; k0 < 96; k0 += 32) {
        v16bf a = load_frag_a(Wbf, mtile * 16, k0, 96);
#pragma unroll
        for (int j = 0; j < 4; ++j)
            acc[j] = wmma_bf16(a, load_frag_b_f32(trow[j], k0), acc[j]);
    }
    int c0 = mtile * 16 + half * 8;
#pragma unroll
    for (int j = 0; j < 4; ++j) {
#pragma unroll
        for (int r = 0; r < 8; ++r) {
            int c = c0 + r;
            float inv = g[c] * rsqrtf(vv[c] + EPSV);
            float val = silu_f(acc[j][r] * inv + (bb[c] - mm[c] * inv));
            size_t oi = ((size_t)(bimg[j] * CH + c)) * HW + yx[j];
            out[oi] = val + x[oi];
        }
    }
}

// ---------------- host ----------------
extern "C" void kernel_launch(void* const* d_in, const int* in_sizes, int n_in,
                              void* d_out, int out_size, void* d_ws, size_t ws_size,
                              hipStream_t stream) {
    const float* x     = (const float*)d_in[0];
    const float* dw_w  = (const float*)d_in[1];
    const float* dw_g  = (const float*)d_in[2];
    const float* dw_b  = (const float*)d_in[3];
    const float* dw_m  = (const float*)d_in[4];
    const float* dw_v  = (const float*)d_in[5];
    const float* pw_w  = (const float*)d_in[6];
    const float* pw_g  = (const float*)d_in[7];
    const float* pw_b  = (const float*)d_in[8];
    const float* pw_m  = (const float*)d_in[9];
    const float* pw_v  = (const float*)d_in[10];
    const float* ln1_g = (const float*)d_in[11];
    const float* ln1_b = (const float*)d_in[12];
    const float* qkv_w = (const float*)d_in[13];
    const float* out_w = (const float*)d_in[14];
    const float* out_b = (const float*)d_in[15];
    const float* ln2_g = (const float*)d_in[16];
    const float* ln2_b = (const float*)d_in[17];
    const float* ff_w1 = (const float*)d_in[18];
    const float* ff_b1 = (const float*)d_in[19];
    const float* ff_w2 = (const float*)d_in[20];
    const float* ff_b2 = (const float*)d_in[21];
    const float* fu_w  = (const float*)d_in[22];
    const float* fu_g  = (const float*)d_in[23];
    const float* fu_b  = (const float*)d_in[24];
    const float* fu_m  = (const float*)d_in[25];
    const float* fu_v  = (const float*)d_in[26];
    float* outp = (float*)d_out;

    char* ws = (char*)d_ws;
    // workspace layout (all offsets 256B aligned)
    __bf16* wbf   = (__bf16*)(ws);                        // 116736 bf16 = 233472 B
    float*  h1    = (float*)(ws + 233472);                // 12.58 MB
    float*  t     = (float*)(ws + 12816384);              // 12.58 MB
    __bf16* xnbf  = (__bf16*)(ws + 25399296);             // 6.29 MB
    float*  qkvb  = (float*)(ws + 31690752);              // 12.58 MB
    __bf16* obf   = (__bf16*)(ws + 44273664);             // 2.10 MB
    __bf16* hbf   = (__bf16*)(ws + 46370816);             // 12.58 MB

    __bf16* pw_bf  = wbf + 0;
    __bf16* qkv_bf = wbf + 9216;
    __bf16* out_bf = wbf + 27648;
    __bf16* ff1_bf = wbf + 33792;
    __bf16* ff2_bf = wbf + 70656;
    __bf16* fu_bf  = wbf + 107520;

    auto cvt = [&](const float* s, __bf16* d, int n) {
        cvt_bf16_kernel<<<(n + 255) / 256, 256, 0, stream>>>(s, d, n);
    };
    cvt(pw_w,  pw_bf,  96 * 96);
    cvt(qkv_w, qkv_bf, 2 * 96 * 96);
    cvt(out_w, out_bf, 2 * 96 * 32);
    cvt(ff_w1, ff1_bf, 2 * 192 * 96);
    cvt(ff_w2, ff2_bf, 2 * 96 * 192);
    cvt(fu_w,  fu_bf,  96 * 96);

    // depthwise conv + BN + SiLU
    dwconv_kernel<<<(BATCH * CH * HW) / 256, 256, 0, stream>>>(
        x, dw_w, dw_g, dw_b, dw_m, dw_v, h1);

    // pointwise conv + BN + SiLU -> token layout: 6 mtiles * 512 ntiles = 3072 waves
    pwconv_kernel<<<384, 256, 0, stream>>>(pw_bf, h1, pw_g, pw_b, pw_m, pw_v, t);

    for (int i = 0; i < 2; ++i) {
        // LN1
        ln_kernel<<<NTOK / 8, 256, 0, stream>>>(t, ln1_g + i * 96, ln1_b + i * 96, xnbf);
        // QKV
        qkv_kernel<<<384, 256, 0, stream>>>(qkv_bf + (size_t)i * 96 * 96, xnbf, qkvb);
        // attention (32 (b,p) groups x 4 heads), 64KB dynamic LDS for K/V
        attn_kernel<<<128, 256, 65536, stream>>>(qkvb, obf);
        // out projection + bias + residual
        proj_kernel<<<384, 256, 0, stream>>>(out_bf + (size_t)i * 96 * 32, obf,
                                             out_b + i * 96, t);
        // LN2
        ln_kernel<<<NTOK / 8, 256, 0, stream>>>(t, ln2_g + i * 96, ln2_b + i * 96, xnbf);
        // FFN: 12 mtiles * 512 ntiles = 6144 waves -> 768 blocks
        ffn1_kernel<<<768, 256, 0, stream>>>(ff1_bf + (size_t)i * 192 * 96, xnbf,
                                             ff_b1 + i * 192, hbf);
        ffn2_kernel<<<384, 256, 0, stream>>>(ff2_bf + (size_t)i * 96 * 192, hbf,
                                             ff_b2 + i * 96, t);
    }

    // fuse conv + BN + SiLU + residual -> output
    fuse_kernel<<<384, 256, 0, stream>>>(fu_bf, t, fu_g, fu_b, fu_m, fu_v, x, outp);
}
